// HybridModel_82222853914741
// MI455X (gfx1250) — matrix operand order repaired and checked by reading, not compiled, not run on hardware
//
#include <hip/hip_runtime.h>
#include <hip/hip_bf16.h>

typedef __attribute__((ext_vector_type(16))) _Float16 v16h;
typedef __attribute__((ext_vector_type(8)))  _Float16 v8h;
typedef __attribute__((ext_vector_type(8)))  float    v8f;

// ---------------------------------------------------------------------------
// Pack input: x f32 NCHW [16,3,224,224] -> f16 haloed NHWC [16,226,226,32]
// ---------------------------------------------------------------------------
__global__ void pack_input_kernel(const float* __restrict__ x, _Float16* __restrict__ out) {
    int idx = blockIdx.x * blockDim.x + threadIdx.x;
    const int total = 16 * 226 * 226 * 32;
    if (idx >= total) return;
    int c = idx & 31;
    int t = idx >> 5;
    int xx = t % 226; t /= 226;
    int yy = t % 226; t /= 226;
    int b = t;
    int h = yy - 1, w = xx - 1;
    float v = 0.0f;
    if (c < 3 && (unsigned)h < 224u && (unsigned)w < 224u)
        v = x[(((size_t)b * 3 + c) * 224 + h) * 224 + w];
    out[idx] = (_Float16)v;
}

// ---------------------------------------------------------------------------
// Pack weights: f32 OIHW [Cout,Cin,3,3] -> f16 [9][Cout][Cin_pad], zero padded
// ---------------------------------------------------------------------------
__global__ void pack_weights_kernel(const float* __restrict__ w, _Float16* __restrict__ wpk,
                                    int Cout, int Cin, int Cin_pad) {
    int idx = blockIdx.x * blockDim.x + threadIdx.x;
    int total = 9 * Cout * Cin_pad;
    if (idx >= total) return;
    int ci = idx % Cin_pad;
    int t  = idx / Cin_pad;
    int co = t % Cout;
    int tap = t / Cout;
    int kh = tap / 3, kw = tap % 3;
    float v = (ci < Cin) ? w[(((size_t)co * Cin + ci) * 3 + kh) * 3 + kw] : 0.0f;
    wpk[idx] = (_Float16)v;
}

// ---------------------------------------------------------------------------
// Zero the 1-pixel halo ring of a haloed NHWC buffer [16, H+2, W+2, C]
// ---------------------------------------------------------------------------
__global__ void zero_halo_kernel(_Float16* __restrict__ buf, int H, int W, int C) {
    int Wp = W + 2, Hp = H + 2;
    int perImg = 2 * Wp * C + 2 * H * C;
    int total = 16 * perImg;
    int idx = blockIdx.x * blockDim.x + threadIdx.x;
    if (idx >= total) return;
    int b = idx / perImg;
    int r = idx % perImg;
    size_t img = (size_t)b * Hp * Wp * C;
    const _Float16 z = (_Float16)0.0f;
    if (r < Wp * C) { buf[img + r] = z; return; }
    r -= Wp * C;
    if (r < Wp * C) { buf[img + (size_t)(Hp - 1) * Wp * C + r] = z; return; }
    r -= Wp * C;
    if (r < H * C) {
        int y = r / C, c = r % C;
        buf[img + (size_t)(y + 1) * Wp * C + c] = z; return;
    }
    r -= H * C;
    {
        int y = r / C, c = r % C;
        buf[img + (size_t)(y + 1) * Wp * C + (size_t)(Wp - 1) * C + c] = z;
    }
}

// ---------------------------------------------------------------------------
// Fragment loader: two 16B loads -> one 16-element f16 operand fragment
// ---------------------------------------------------------------------------
__device__ __forceinline__ v16h load_frag(const _Float16* p, int off2) {
    v8h lo = *(const v8h*)(p);
    v8h hi = *(const v8h*)(p + off2);
    return __builtin_shufflevector(lo, hi, 0,1,2,3,4,5,6,7,8,9,10,11,12,13,14,15);
}

// ---------------------------------------------------------------------------
// Implicit-GEMM conv3x3 SAME on haloed NHWC f16, WMMA f32_16x16x32_f16,
// fused bias+ReLU.
//  - 128-thread block (4 waves) shares one 32-channel Cout group.
//  - Per tap: block async-copies the contiguous 32 x Cin_pad f16 weight tile
//    into LDS (global_load_async_to_lds_b128, ASYNCcnt), barriers, then each
//    wave runs the K-loop with A-frags from LDS and B-frags from global.
//  - Each wave: 2 co-tiles x 4 pos-tiles -> 8 WMMA per 32-wide K chunk.
//  - All loads unconditional (halo + clamp); only stores are masked, so EXEC
//    is all-ones for every WMMA.
// ---------------------------------------------------------------------------
__global__ __launch_bounds__(128) void conv3x3_wmma_kernel(
    const _Float16* __restrict__ in,   // [B, H+2, W+2, Cin_pad] haloed
    const _Float16* __restrict__ wpk,  // [9, Cout, Cin_pad]
    const float*    __restrict__ bias, // [Cout]
    _Float16*       __restrict__ out,  // [B, H+2, W+2, Cout] haloed
    int H, int W, int Cin_pad, int Cout, int Bn)
{
    extern __shared__ __align__(16) _Float16 sA[];   // [32][Cin_pad + 8] (row pad vs bank conflicts)
    const int ldsRowH = Cin_pad + 8;

    const int lane = threadIdx.x & 31;
    const int wave = threadIdx.x >> 5;
    const int HW = H * W;
    const int posG  = (HW + 63) >> 6;          // 64-position groups
    const int posBG = (posG + 3) >> 2;         // 4 groups per block
    const int coG   = Cout >> 5;

    int blk = blockIdx.x;
    int co_g = blk % coG;  int t2 = blk / coG;
    int pos_b = t2 % posBG; int b = t2 / posBG;
    int pos_g = pos_b * 4 + wave;

    const size_t rowI = (size_t)(W + 2) * Cin_pad;
    const size_t imgI = (size_t)(H + 2) * rowI;
    const size_t rowO = (size_t)(W + 2) * Cout;
    const size_t imgO = (size_t)(H + 2) * rowO;

    // operand lane striping
    const int m      = lane & 15;
    const int khalfA = (lane >> 4) << 4;       // 0 or 16
    const int n      = lane & 15;
    const int kbaseB = (lane >> 4) << 4;       // 0 or 16

    int  hj[4], wj[4];
    bool pv[4];
    #pragma unroll
    for (int j = 0; j < 4; j++) {
        int p = pos_g * 64 + j * 16 + n;
        pv[j] = (p < HW);
        int pc = pv[j] ? p : (HW - 1);
        hj[j] = pc / W;
        wj[j] = pc % W;
    }

    v8f acc[2][4];
    #pragma unroll
    for (int cj = 0; cj < 2; cj++)
        #pragma unroll
        for (int pj = 0; pj < 4; pj++)
            #pragma unroll
            for (int i = 0; i < 8; i++) acc[cj][pj][i] = 0.0f;

    const _Float16* inb = in + (size_t)b * imgI;
    const uint32_t ldsBase = (uint32_t)(uintptr_t)sA;
    const int rowChunks = Cin_pad >> 3;        // 16B chunks per weight row
    const int tileChunks = 4 * Cin_pad;        // 32 rows * rowChunks

    // LDS A-frag base pointers for this lane
    const _Float16* lp0 = sA + (size_t)m * ldsRowH + khalfA;
    const _Float16* lp1 = sA + (size_t)(16 + m) * ldsRowH + khalfA;

    #pragma unroll
    for (int tap = 0; tap < 9; tap++) {
        const int kh = tap / 3, kw = tap % 3;
        // weight tile for (tap, co_g): contiguous 32*Cin_pad halves in global
        const _Float16* gsrc = wpk + ((size_t)tap * Cout + (size_t)co_g * 32) * Cin_pad;

        __syncthreads();                        // previous tap's LDS reads done
        for (int idx = threadIdx.x; idx < tileChunks; idx += 128) {
            int row = idx / rowChunks;
            int cc  = idx - row * rowChunks;
            uint32_t dsto = ldsBase + (uint32_t)(row * ldsRowH + cc * 8) * 2u;
            const void* src = (const void*)(gsrc + (size_t)idx * 8);
            asm volatile("global_load_async_to_lds_b128 %0, %1, off"
                         :: "v"(dsto), "v"(src) : "memory");
        }
        asm volatile("s_wait_asynccnt 0x0" ::: "memory");
        __syncthreads();                        // tile visible to all waves

        const _Float16* bp[4];
        #pragma unroll
        for (int j = 0; j < 4; j++)
            bp[j] = inb + (size_t)(hj[j] + kh) * rowI + (size_t)(wj[j] + kw) * Cin_pad + kbaseB;
        __builtin_prefetch(bp[0] + rowI, 0, 1);

        for (int kc = 0; kc < Cin_pad; kc += 32) {
            v16h a0 = load_frag(lp0 + kc, 8);  // ds_load_b128 x2
            v16h a1 = load_frag(lp1 + kc, 8);
            v16h b0 = load_frag(bp[0] + kc, 8);
            v16h b1 = load_frag(bp[1] + kc, 8);
            v16h b2 = load_frag(bp[2] + kc, 8);
            v16h b3 = load_frag(bp[3] + kc, 8);
            acc[0][0] = __builtin_amdgcn_wmma_f32_16x16x32_f16(false, a0, false, b0, (short)0, acc[0][0], false, false);
            acc[0][1] = __builtin_amdgcn_wmma_f32_16x16x32_f16(false, a0, false, b1, (short)0, acc[0][1], false, false);
            acc[0][2] = __builtin_amdgcn_wmma_f32_16x16x32_f16(false, a0, false, b2, (short)0, acc[0][2], false, false);
            acc[0][3] = __builtin_amdgcn_wmma_f32_16x16x32_f16(false, a0, false, b3, (short)0, acc[0][3], false, false);
            acc[1][0] = __builtin_amdgcn_wmma_f32_16x16x32_f16(false, a1, false, b0, (short)0, acc[1][0], false, false);
            acc[1][1] = __builtin_amdgcn_wmma_f32_16x16x32_f16(false, a1, false, b1, (short)0, acc[1][1], false, false);
            acc[1][2] = __builtin_amdgcn_wmma_f32_16x16x32_f16(false, a1, false, b2, (short)0, acc[1][2], false, false);
            acc[1][3] = __builtin_amdgcn_wmma_f32_16x16x32_f16(false, a1, false, b3, (short)0, acc[1][3], false, false);
        }
    }

    // Epilogue: D layout lane = N(pos), VGPR r -> M = r + 8*(lane>=16)
    const int mbase = (lane >> 4) << 3;
    _Float16* outb = out + (size_t)b * imgO;
    #pragma unroll
    for (int cj = 0; cj < 2; cj++) {
        const int co0 = co_g * 32 + cj * 16 + mbase;
        float bs[8];
        #pragma unroll
        for (int r = 0; r < 8; r++) bs[r] = bias[co0 + r];
        #pragma unroll
        for (int pj = 0; pj < 4; pj++) {
            v8h o;
            #pragma unroll
            for (int r = 0; r < 8; r++)
                o[r] = (_Float16)fmaxf(acc[cj][pj][r] + bs[r], 0.0f);
            if (pv[pj])
                *(v8h*)(outb + (size_t)(hj[pj] + 1) * rowO + (size_t)(wj[pj] + 1) * Cout + co0) = o;
        }
    }
}

// ---------------------------------------------------------------------------
// 2x2/2 maxpool on haloed NHWC f16 buffers
// ---------------------------------------------------------------------------
__global__ void maxpool_kernel(const _Float16* __restrict__ in, _Float16* __restrict__ out,
                               int Ho, int Wo, int C) {
    int idx = blockIdx.x * blockDim.x + threadIdx.x;
    int total = 16 * Ho * Wo * C;
    if (idx >= total) return;
    int c = idx % C; int t = idx / C;
    int wo = t % Wo; t /= Wo;
    int ho = t % Ho; int b = t / Ho;
    const int Hi = 2 * Ho, Wi = 2 * Wo;
    const size_t rowI = (size_t)(Wi + 2) * C;
    const size_t imgI = (size_t)(Hi + 2) * rowI;
    const size_t rowO = (size_t)(Wo + 2) * C;
    const size_t imgO = (size_t)(Ho + 2) * rowO;
    const _Float16* p0 = in + (size_t)b * imgI + (size_t)(2 * ho + 1) * rowI + (size_t)(2 * wo + 1) * C + c;
    float a  = (float)p0[0];
    float bb = (float)p0[C];
    float cc = (float)p0[rowI];
    float dd = (float)p0[rowI + C];
    out[(size_t)b * imgO + (size_t)(ho + 1) * rowO + (size_t)(wo + 1) * C + c] =
        (_Float16)fmaxf(fmaxf(a, bb), fmaxf(cc, dd));
}

// ---------------------------------------------------------------------------
// Global average pool: haloed [16,9,9,512] f16 interior -> [16,512] f32
// ---------------------------------------------------------------------------
__global__ void avgpool_kernel(const _Float16* __restrict__ in, float* __restrict__ feat) {
    int idx = blockIdx.x * blockDim.x + threadIdx.x;
    if (idx >= 16 * 512) return;
    int c = idx & 511, b = idx >> 9;
    const _Float16* p = in + (size_t)b * 81 * 512 + c;
    float s = 0.0f;
    #pragma unroll
    for (int y = 0; y < 7; y++)
        #pragma unroll
        for (int x = 0; x < 7; x++)
            s += (float)p[(size_t)((y + 1) * 9 + (x + 1)) * 512];
    feat[idx] = s * (1.0f / 49.0f);
}

// ---------------------------------------------------------------------------
// 4-qubit statevector sim (RY encode, Rot layer, CNOT ring r=1, <Z_i>)
// ---------------------------------------------------------------------------
__device__ void qsim4(const float* in4, const float* qw, float* exps) {
    float sr[16], si[16];
    for (int s = 0; s < 16; s++) { sr[s] = 0.0f; si[s] = 0.0f; }
    sr[0] = 1.0f;
    for (int q = 0; q < 4; q++) {
        float c = cosf(0.5f * in4[q]), s = sinf(0.5f * in4[q]);
        int mk = 1 << (3 - q);
        for (int idx = 0; idx < 16; idx++) {
            if (idx & mk) continue;
            int j = idx | mk;
            float ar = sr[idx], ai = si[idx], br = sr[j], bi = si[j];
            sr[idx] = c * ar - s * br;  si[idx] = c * ai - s * bi;
            sr[j]   = s * ar + c * br;  si[j]   = s * ai + c * bi;
        }
    }
    for (int q = 0; q < 4; q++) {
        float phi = qw[q * 3 + 0], th = qw[q * 3 + 1], om = qw[q * 3 + 2];
        float c = cosf(0.5f * th), s = sinf(0.5f * th);
        float apo = 0.5f * (phi + om), amo = 0.5f * (phi - om);
        float u00r =  cosf(apo) * c, u00i = -sinf(apo) * c;
        float u01r = -cosf(amo) * s, u01i = -sinf(amo) * s;
        float u10r =  cosf(amo) * s, u10i = -sinf(amo) * s;
        float u11r =  cosf(apo) * c, u11i =  sinf(apo) * c;
        int mk = 1 << (3 - q);
        for (int idx = 0; idx < 16; idx++) {
            if (idx & mk) continue;
            int j = idx | mk;
            float ar = sr[idx], ai = si[idx], br = sr[j], bi = si[j];
            sr[idx] = u00r * ar - u00i * ai + u01r * br - u01i * bi;
            si[idx] = u00r * ai + u00i * ar + u01r * bi + u01i * br;
            sr[j]   = u10r * ar - u10i * ai + u11r * br - u11i * bi;
            si[j]   = u10r * ai + u10i * ar + u11r * bi + u11i * br;
        }
    }
    for (int q = 0; q < 4; q++) {
        int mc = 1 << (3 - q), mt = 1 << (3 - ((q + 1) & 3));
        for (int idx = 0; idx < 16; idx++) {
            if ((idx & mc) && !(idx & mt)) {
                int j = idx | mt;
                float tr = sr[idx], ti = si[idx];
                sr[idx] = sr[j]; si[idx] = si[j];
                sr[j] = tr;      si[j] = ti;
            }
        }
    }
    for (int q = 0; q < 4; q++) {
        int mk = 1 << (3 - q);
        float e = 0.0f;
        for (int s = 0; s < 16; s++) {
            float pr = sr[s] * sr[s] + si[s] * si[s];
            e += (s & mk) ? -pr : pr;
        }
        exps[q] = e;
    }
}

// ---------------------------------------------------------------------------
// Head: dense(512->128)+relu, dense(128->4)+tanh, qcircuit residual,
//       dense(4->32)+relu, dense(32->2). One block per batch element.
// ---------------------------------------------------------------------------
__global__ __launch_bounds__(128) void head_kernel(
    const float* __restrict__ feat,
    const float* __restrict__ w1, const float* __restrict__ b1,
    const float* __restrict__ w2, const float* __restrict__ b2,
    const float* __restrict__ qw,
    const float* __restrict__ fw1, const float* __restrict__ fb1,
    const float* __restrict__ fw2, const float* __restrict__ fb2,
    float* __restrict__ out)
{
    __shared__ float h1[128];
    __shared__ float h2s[4];
    __shared__ float h3[4];
    __shared__ float h4[32];
    int b = blockIdx.x, t = threadIdx.x;
    const float* f = feat + (size_t)b * 512;

    float a = b1[t];
    for (int k = 0; k < 512; k++) a += w1[(size_t)t * 512 + k] * f[k];
    h1[t] = fmaxf(a, 0.0f);
    __syncthreads();

    if (t < 4) {
        float s = b2[t];
        for (int k = 0; k < 128; k++) s += w2[t * 128 + k] * h1[k];
        h2s[t] = tanhf(s);
    }
    __syncthreads();

    if (t == 0) {
        float qr[4];
        float in4[4] = {h2s[0], h2s[1], h2s[2], h2s[3]};
        qsim4(in4, qw, qr);
        for (int i = 0; i < 4; i++) h3[i] = in4[i] + 0.1f * qr[i];
    }
    __syncthreads();

    if (t < 32) {
        float s = fb1[t];
        for (int k = 0; k < 4; k++) s += fw1[t * 4 + k] * h3[k];
        h4[t] = fmaxf(s, 0.0f);
    }
    __syncthreads();

    if (t < 2) {
        float s = fb2[t];
        for (int k = 0; k < 32; k++) s += fw2[t * 32 + k] * h4[k];
        out[b * 2 + t] = s;
    }
}

// ---------------------------------------------------------------------------
// Launcher
// d_in: 0:x, 1..26: (w_i,b_i) x13, 27:dnn_w1, 28:dnn_b1, 29:dnn_w2, 30:dnn_b2,
//       31:q_weights, 32:fc_w1, 33:fc_b1, 34:fc_w2, 35:fc_b2
// ---------------------------------------------------------------------------
extern "C" void kernel_launch(void* const* d_in, const int* in_sizes, int n_in,
                              void* d_out, int out_size, void* d_ws, size_t ws_size,
                              hipStream_t stream) {
    (void)in_sizes; (void)n_in; (void)out_size; (void)ws_size;
    const float* x = (const float*)d_in[0];
    static const int couts[13] = {64,64,128,128,256,256,256,512,512,512,512,512,512};
    static const int pools[13] = {0,1,0,1,0,0,1,0,0,1,0,0,1};

    char* ws = (char*)d_ws;
    const size_t ACT_ELEMS = (size_t)16 * 226 * 226 * 64;   // max haloed NHWC activation
    _Float16* actA = (_Float16*)ws;
    _Float16* actB = (_Float16*)(ws + ACT_ELEMS * sizeof(_Float16));
    char* wbase = ws + 2 * ACT_ELEMS * sizeof(_Float16);

    size_t woff[13]; size_t wtot = 0;
    {
        int cin_pad = 32;
        for (int i = 0; i < 13; i++) {
            woff[i] = wtot;
            wtot += (size_t)9 * couts[i] * cin_pad;
            cin_pad = couts[i];
        }
    }
    _Float16* wpk = (_Float16*)wbase;
    float* feat = (float*)(wbase + wtot * sizeof(_Float16));

    {
        int total = 16 * 226 * 226 * 32;
        pack_input_kernel<<<(total + 255) / 256, 256, 0, stream>>>(x, actA);
    }
    {
        int cin = 3, cin_pad = 32;
        for (int i = 0; i < 13; i++) {
            int total = 9 * couts[i] * cin_pad;
            pack_weights_kernel<<<(total + 255) / 256, 256, 0, stream>>>(
                (const float*)d_in[1 + 2 * i], wpk + woff[i], couts[i], cin, cin_pad);
            cin = couts[i]; cin_pad = cin;
        }
    }
    _Float16* cur = actA; _Float16* nxt = actB;
    int H = 224;
    int cin_pad = 32;
    for (int i = 0; i < 13; i++) {
        int cout = couts[i];
        {
            int total = 16 * (2 * (H + 2) * cout + 2 * H * cout);
            zero_halo_kernel<<<(total + 255) / 256, 256, 0, stream>>>(nxt, H, H, cout);
        }
        int HW = H * H;
        int posG = (HW + 63) / 64;
        int posBG = (posG + 3) / 4;
        int blocks = 16 * posBG * (cout / 32);
        size_t ldsBytes = (size_t)32 * (cin_pad + 8) * sizeof(_Float16);
        conv3x3_wmma_kernel<<<blocks, 128, ldsBytes, stream>>>(
            cur, wpk + woff[i], (const float*)d_in[2 + 2 * i], nxt,
            H, H, cin_pad, cout, 16);
        { _Float16* t = cur; cur = nxt; nxt = t; }
        if (pools[i]) {
            int Ho = H / 2;
            {
                int total = 16 * (2 * (Ho + 2) * cout + 2 * Ho * cout);
                zero_halo_kernel<<<(total + 255) / 256, 256, 0, stream>>>(nxt, Ho, Ho, cout);
            }
            int total = 16 * Ho * Ho * cout;
            maxpool_kernel<<<(total + 255) / 256, 256, 0, stream>>>(cur, nxt, Ho, Ho, cout);
            { _Float16* t = cur; cur = nxt; nxt = t; }
            H = Ho;
        }
        cin_pad = cout;
    }
    avgpool_kernel<<<(16 * 512 + 255) / 256, 256, 0, stream>>>(cur, feat);
    head_kernel<<<16, 128, 0, stream>>>(feat,
        (const float*)d_in[27], (const float*)d_in[28],
        (const float*)d_in[29], (const float*)d_in[30],
        (const float*)d_in[31],
        (const float*)d_in[32], (const float*)d_in[33],
        (const float*)d_in[34], (const float*)d_in[35],
        (float*)d_out);
}
